// SSIM_15461882265932
// MI455X (gfx1250) — matrix operand order repaired and checked by reading, not compiled, not run on hardware
//
#include <hip/hip_runtime.h>

typedef __bf16 bf16_t;
typedef bf16_t v16bf __attribute__((ext_vector_type(16)));
typedef bf16_t v8bf  __attribute__((ext_vector_type(8)));
typedef bf16_t v4bf  __attribute__((ext_vector_type(4)));
typedef float  v8f   __attribute__((ext_vector_type(8)));
typedef float  v4f   __attribute__((ext_vector_type(4)));

#define SSIM_C1 1.0e-4f
#define SSIM_C2 9.0e-4f

#define SROW   88            // bf16 row stride of staged planes (80 cols + pad)
#define SPLANE (32 * SROW)   // elements per staged source plane
#define HROW   40            // bf16 row stride of transposed H buffer

// 11-tap Gaussian (sigma=1.5), idx in [0,10]; zero outside band.
// NOTE: multiply by 1/4.5 (constant) -- never a runtime fp32 divide.
__device__ __forceinline__ float gw(int idx, float invsum) {
    float d = (float)(idx - 5);
    float w = __expf(d * d * -0.222222222f) * invsum;
    return (idx < 0 || idx > 10) ? 0.0f : w;
}

// Fast fp32 reciprocal: v_rcp_f32 + one Newton-Raphson step (~0.5 ulp).
__device__ __forceinline__ float fast_rcp(float x) {
    float r = __builtin_amdgcn_rcpf(x);
    return r * (2.0f - x * r);
}

// One wave computes FOUR x-adjacent 16x16 output tiles from one shared
// 32x80 staged halo. blockDim = 64 -> 2 independent waves per block.
__global__ void __launch_bounds__(64) ssim_wmma_kernel(
    const float* __restrict__ img1, const float* __restrict__ img2,
    float* __restrict__ partial)
{
    __shared__ bf16_t stg[2][5 * SPLANE];  // 5 bf16 source planes, 32 rows x 80 cols
    __shared__ bf16_t hts[2][16 * HROW];   // Ht[n][k]: transposed horizontal result (bf16)
    __shared__ float  bsum[2];

    const int wave = threadIdx.x >> 5;
    const int lane = threadIdx.x & 31;
    const int grp  = blockIdx.x * 2 + wave;  // group of 4 x-adjacent tiles
    const int plane = grp >> 8;              // 256 groups per 512x512 plane
    const int rem   = grp & 255;
    const int ty  = rem >> 3;                // 0..31
    const int txg = rem & 7;                 // 0..7
    const int y0 = ty * 16;
    const int xb = txg * 64;                 // x0 of first tile in group

    const float* __restrict__ p1 = img1 + (size_t)plane * 262144u;
    const float* __restrict__ p2 = img2 + (size_t)plane * 262144u;
    bf16_t* S  = stg[wave];
    bf16_t* HT = hts[wave];

    // ---- stage rows [y0-8,y0+24) x cols [xb-8,xb+72) as 5 bf16 planes ----
    // products formed in f32 from freshly loaded values, rounded once.
    {
        int yr = y0 - 8 + lane;              // one image row per lane
        bool yok = (yr >= 0) && (yr < 512);
        const float* r1 = p1 + yr * 512;
        const float* r2 = p2 + yr * 512;
        bf16_t* row = S + lane * SROW;
        #pragma unroll
        for (int c = 0; c < 20; ++c) {
            int xs = xb - 8 + c * 4;         // 4-aligned chunk: fully in or fully out
            v4f a = {0.f, 0.f, 0.f, 0.f};
            v4f b = {0.f, 0.f, 0.f, 0.f};
            if (yok && xs >= 0 && xs < 512) {
                a = *(const v4f*)(r1 + xs);
                b = *(const v4f*)(r2 + xs);
            }
            v4f p11 = a * a, p22 = b * b, p12 = a * b;
            *(v4bf*)(row + 0 * SPLANE + c * 4) = __builtin_convertvector(a,   v4bf);
            *(v4bf*)(row + 1 * SPLANE + c * 4) = __builtin_convertvector(b,   v4bf);
            *(v4bf*)(row + 2 * SPLANE + c * 4) = __builtin_convertvector(p11, v4bf);
            *(v4bf*)(row + 3 * SPLANE + c * 4) = __builtin_convertvector(p22, v4bf);
            *(v4bf*)(row + 4 * SPLANE + c * 4) = __builtin_convertvector(p12, v4bf);
        }
    }

    // ---- Gaussian weight operands: built ONCE per wave (amortized over 4 tiles) ----
    float wsum = 0.f;
    #pragma unroll
    for (int i = 0; i < 11; ++i) {
        float d = (float)(i - 5);
        wsum += __expf(d * d * -0.222222222f);
    }
    const float inv = fast_rcp(wsum);

    const int n    = lane & 15;          // B/C column, A row
    const int khi  = lane & 16;          // B operand K offset (0 / 16)
    const int kb   = (lane & 16) >> 1;   // A operand K base   (0 / 8)
    const int mofs = kb;                 // C-layout row offset (0 / 8)

    v16bf whb;  // horizontal weights as B (32x16): B[k][n] = g[k - n - 3]
    #pragma unroll
    for (int h = 0; h < 16; ++h)
        whb[h] = (bf16_t)gw(h + kb + ((h >= 8) ? 8 : 0) - n - 3, inv);

    v16bf wva;  // vertical weights as A (16x32): A[r][k] = g[k - r - 3]
    #pragma unroll
    for (int i = 0; i < 16; ++i) {
        int K = khi + i;
        wva[i] = (bf16_t)gw(K - n - 3, inv);
    }

    const v8f z8 = {0.f, 0.f, 0.f, 0.f, 0.f, 0.f, 0.f, 0.f};
    float acc = 0.f;

    for (int t = 0; t < 4; ++t) {
        const int cb = t * 16;  // staged-column base of this tile (K=0 <-> col cb)

        // ---- horizontal pass: A operands come straight from bf16 LDS ----
        v8f Hac[5][2];
        #pragma unroll
        for (int s = 0; s < 5; ++s) {
            #pragma unroll
            for (int g = 0; g < 2; ++g) {
                const bf16_t* R = S + s * SPLANE + (g * 16 + n) * SROW + cb + kb;
                v8bf lo = *(const v8bf*)(R);        // K = kb .. kb+7
                v8bf hi = *(const v8bf*)(R + 16);   // K = kb+16 .. kb+23
                v16bf A = __builtin_shufflevector(lo, hi,
                    0,1,2,3,4,5,6,7,8,9,10,11,12,13,14,15);
                Hac[s][g] = __builtin_amdgcn_wmma_f32_16x16x32_bf16(
                    false, A, false, whb, (short)0, z8, false, false);
            }
        }

        // ---- vertical pass per source: H -> bf16 Ht -> B operand -> WMMA ----
        v8f Out[5];
        #pragma unroll
        for (int s = 0; s < 5; ++s) {
            #pragma unroll
            for (int g = 0; g < 2; ++g) {
                v4f flo = __builtin_shufflevector(Hac[s][g], Hac[s][g], 0, 1, 2, 3);
                v4f fhi = __builtin_shufflevector(Hac[s][g], Hac[s][g], 4, 5, 6, 7);
                bf16_t* W = HT + n * HROW + g * 16 + mofs;   // k = g*16 + mofs + j
                *(v4bf*)(W)     = __builtin_convertvector(flo, v4bf);
                *(v4bf*)(W + 4) = __builtin_convertvector(fhi, v4bf);
            }
            const bf16_t* HR = HT + n * HROW + khi;
            v8bf blo = *(const v8bf*)(HR);       // K = khi .. khi+7
            v8bf bhi = *(const v8bf*)(HR + 8);   // K = khi+8 .. khi+15
            v16bf B = __builtin_shufflevector(blo, bhi,
                0,1,2,3,4,5,6,7,8,9,10,11,12,13,14,15);
            Out[s] = __builtin_amdgcn_wmma_f32_16x16x32_bf16(
                false, wva, false, B, (short)0, z8, false, false);
        }

        // ---- elementwise SSIM, vectorized over the v8f accumulators ----
        {
            v8f mu1 = Out[0], mu2 = Out[1];
            v8f m11 = mu1 * mu1, m22 = mu2 * mu2, m12 = mu1 * mu2;
            v8f s11 = Out[2] - m11;
            v8f s22 = Out[3] - m22;
            v8f s12 = Out[4] - m12;
            v8f num = (2.0f * m12 + SSIM_C1) * (2.0f * s12 + SSIM_C2);
            v8f den = (m11 + m22 + SSIM_C1) * (s11 + s22 + SSIM_C2);
            #pragma unroll
            for (int j = 0; j < 8; ++j)
                acc += num[j] * fast_rcp(den[j]);
        }
    }

    // ---- reduction: wave shfl tree, then per-block partial ----
    #pragma unroll
    for (int off = 16; off > 0; off >>= 1)
        acc += __shfl_xor(acc, off, 32);
    if (lane == 0) bsum[wave] = acc;
    __syncthreads();
    if (threadIdx.x == 0)
        partial[blockIdx.x] = bsum[0] + bsum[1];
}

// Deterministic final reduction: fixed tree over block partials.
__global__ void __launch_bounds__(256) ssim_reduce_kernel(
    const float* __restrict__ partial, int nb, float* __restrict__ out)
{
    __shared__ float sm[256];
    float a = 0.f;
    for (int i = threadIdx.x; i < nb; i += 256) a += partial[i];
    sm[threadIdx.x] = a;
    __syncthreads();
    for (int s = 128; s > 0; s >>= 1) {
        if (threadIdx.x < s) sm[threadIdx.x] += sm[threadIdx.x + s];
        __syncthreads();
    }
    if (threadIdx.x == 0) out[0] = sm[0] * (1.0f / 25165824.0f); // 32*3*512*512
}

extern "C" void kernel_launch(void* const* d_in, const int* in_sizes, int n_in,
                              void* d_out, int out_size, void* d_ws, size_t ws_size,
                              hipStream_t stream) {
    const float* img1 = (const float*)d_in[0];
    const float* img2 = (const float*)d_in[1];
    float* partial = (float*)d_ws;            // 12288 floats = 48 KB scratch

    const int nGroups = 96 * 32 * 8;          // planes * (32 ty) * (8 groups of 4 tiles)
    const int nBlocks = nGroups / 2;          // 2 waves (groups) per block -> 12288

    ssim_wmma_kernel<<<nBlocks, 64, 0, stream>>>(img1, img2, partial);
    ssim_reduce_kernel<<<1, 256, 0, stream>>>(partial, nBlocks, (float*)d_out);
}
